// Cross_Attention_25786983645651
// MI455X (gfx1250) — matrix operand order repaired and checked by reading, not compiled, not run on hardware
//
#include <hip/hip_runtime.h>

// ---------------------------------------------------------------------------
// Cross-attention (efficient/linear attention) + 1x1 conv + LayerNorm, MI455X.
//   y[b,n,o] = LN_o( sum_c W2_b[o,c] * x1[b,n,c] + bias[o] )
//   W2_b = W * blockdiag_h(S_b),  S_b[h] = softmax_c(kq) @ softmax_N(kq)^T
// Heavy GEMM done with v_wmma_f32_16x16x32_f16 (f32 accumulate); everything
// else fp32 VALU. All scratch in d_ws (~37 MiB). Deterministic (no atomics).
// ---------------------------------------------------------------------------

typedef __attribute__((ext_vector_type(16))) _Float16     v16h;
typedef __attribute__((ext_vector_type(4)))  _Float16     h4v;
typedef __attribute__((ext_vector_type(8)))  float        v8f;
typedef __attribute__((ext_vector_type(4)))  unsigned int u32x4;

#define BB 4
#define NN 4096
#define CC 1024
#define HH 8
#define CH 128
#define OO 2048
#define SCHUNKS 8                // token chunks for S partials
#define STOK    (NN / SCHUNKS)   // 512 tokens per S chunk

union FragH { v16h v; u32x4 q[2]; };

// ---------------- key softmax stats: max/sum over tokens per (b, channel) ---
__global__ __launch_bounds__(256) void colstats_kernel(
    const float* __restrict__ x2, float* __restrict__ kmax, float* __restrict__ ksum) {
  int t = blockIdx.x * 256 + threadIdx.x;       // 0..4095  (b*1024 + ch)
  int b = t >> 10, ch = t & 1023;
  const float* p = x2 + (size_t)b * NN * CC + ch;
  float m = -3.4e38f, s = 0.f;
  for (int n = 0; n < NN; ++n) {                // online softmax accumulation
    float v = p[(size_t)n * CC];
    if (v > m) { s = s * __expf(m - v) + 1.f; m = v; }
    else       { s += __expf(v - m); }
  }
  kmax[t] = m; ksum[t] = s;
}

// ---------------- query softmax stats: max/sum over 128 channels per (b,h,n)
__global__ __launch_bounds__(256) void qstats_kernel(
    const float* __restrict__ x2, float* __restrict__ qmax, float* __restrict__ qsum) {
  int t = blockIdx.x * 256 + threadIdx.x;       // 0..131071
  int b = t >> 15;
  int n = (t >> 3) & 4095;
  int h = t & 7;
  const float4* p = (const float4*)(x2 + ((size_t)(b * NN + n)) * CC + h * CH);
  float m = -3.4e38f;
  for (int i = 0; i < 32; ++i) {
    float4 v = p[i];
    m = fmaxf(m, fmaxf(fmaxf(v.x, v.y), fmaxf(v.z, v.w)));
  }
  float s = 0.f;
  for (int i = 0; i < 32; ++i) {
    float4 v = p[i];
    s += __expf(v.x - m) + __expf(v.y - m) + __expf(v.z - m) + __expf(v.w - m);
  }
  qmax[(b * HH + h) * NN + n] = m;
  qsum[(b * HH + h) * NN + n] = s;
}

// ---------------- S partials: S[b,h,i,j] += sum_{n in chunk} q[i,n]*k[j,n] ---
__global__ __launch_bounds__(256) void smat_kernel(
    const float* __restrict__ x2,
    const float* __restrict__ kmax, const float* __restrict__ ksum,
    const float* __restrict__ qmax, const float* __restrict__ qsum,
    float* __restrict__ Spart) {
  __shared__ float qs[8][132];
  __shared__ float ks[8][132];
  int bid = blockIdx.x;                 // 256 blocks = b(4) * h(8) * chunk(8)
  int nc = bid & 7;
  int h  = (bid >> 3) & 7;
  int b  = bid >> 6;
  int tid = threadIdx.x;
  int ti = tid >> 4, tj = tid & 15;     // 16x16 thread grid over 128x128 S
  float acc[8][8];
#pragma unroll
  for (int a = 0; a < 8; ++a)
#pragma unroll
    for (int c = 0; c < 8; ++c) acc[a][c] = 0.f;

  int n0 = nc * STOK;
  const float* kmx = kmax + b * CC + h * CH;
  const float* ksm = ksum + b * CC + h * CH;
  const float* qmx = qmax + (b * HH + h) * NN;
  const float* qsm = qsum + (b * HH + h) * NN;

  for (int st = 0; st < STOK / 8; ++st) {
    for (int idx = tid; idx < 8 * CH; idx += 256) {   // stage 8 tokens
      int tok = idx >> 7, i = idx & 127;
      int n = n0 + st * 8 + tok;
      float v = x2[((size_t)(b * NN + n)) * CC + h * CH + i];
      qs[tok][i] = __expf(v - qmx[n]) / qsm[n];
      ks[tok][i] = __expf(v - kmx[i]) / ksm[i];
    }
    __syncthreads();
#pragma unroll
    for (int tok = 0; tok < 8; ++tok) {
      float4 q0 = *(const float4*)&qs[tok][ti * 8];
      float4 q1 = *(const float4*)&qs[tok][ti * 8 + 4];
      float4 k0 = *(const float4*)&ks[tok][tj * 8];
      float4 k1 = *(const float4*)&ks[tok][tj * 8 + 4];
      float qv[8] = {q0.x, q0.y, q0.z, q0.w, q1.x, q1.y, q1.z, q1.w};
      float kv[8] = {k0.x, k0.y, k0.z, k0.w, k1.x, k1.y, k1.z, k1.w};
#pragma unroll
      for (int a = 0; a < 8; ++a)
#pragma unroll
        for (int c = 0; c < 8; ++c)
          acc[a][c] = fmaf(qv[a], kv[c], acc[a][c]);
    }
    __syncthreads();
  }
  float* Sp = Spart + ((size_t)nc * (BB * HH) + (b * HH + h)) * (CH * CH);
#pragma unroll
  for (int a = 0; a < 8; ++a)
#pragma unroll
    for (int c = 0; c < 8; ++c)
      Sp[(ti * 8 + a) * CH + (tj * 8 + c)] = acc[a][c];
}

// ---------------- deterministic reduction of S partials ---------------------
__global__ __launch_bounds__(256) void sreduce_kernel(
    const float* __restrict__ Spart, float* __restrict__ S) {
  int t = blockIdx.x * 256 + threadIdx.x;  // 524288 elements of S
  float s = 0.f;
  for (int c = 0; c < SCHUNKS; ++c)
    s += Spart[(size_t)c * (BB * HH * CH * CH) + t];
  S[t] = s;
}

// ---------------- fold S into conv weight: W2[b,o,c'] -> f16 -----------------
__global__ __launch_bounds__(256) void w2_kernel(
    const float* __restrict__ w_proj, const float* __restrict__ S,
    _Float16* __restrict__ W2) {
  size_t t = (size_t)blockIdx.x * 256 + threadIdx.x;  // B*O*C = 8,388,608
  int cp = (int)(t & 1023);
  int o  = (int)((t >> 10) & 2047);
  int b  = (int)(t >> 21);
  int h = cp >> 7, j = cp & 127;
  const float* wrow = w_proj + (size_t)o * CC + h * CH;
  const float* Sp   = S + ((size_t)(b * HH + h)) * (CH * CH) + j;
  float acc = 0.f;
  for (int i = 0; i < CH; ++i)
    acc = fmaf(wrow[i], Sp[(size_t)i * CH], acc);
  W2[t] = (_Float16)acc;
}

// ---------------- main GEMM: y[b,n,o] = W2_b @ x1_b + bias (WMMA f16) --------
// block tile 256(o) x 256(n), K = 1024; 16 waves, wave owns one 16-row o-band
// and all 16 n-tiles -> 128 accumulator VGPRs, A-fragment reused 16x.
// B-fragment loads software-pipelined 2 tiles ahead (3 rotating buffers);
// A fragment for the next k-step prefetched during the current compute phase.
__global__ __launch_bounds__(512) void gemm_kernel(
    const _Float16* __restrict__ W2, const float* __restrict__ x1,
    const float* __restrict__ bias, float* __restrict__ y) {
  __shared__ _Float16 sb[256][40];        // V tile [n][k] f16, padded rows
  __shared__ float tsc[16][16][17];       // per-wave transpose scratch
  int b  = blockIdx.z;
  int o0 = blockIdx.y * 256;
  int n0 = blockIdx.x * 256;
  int tid  = threadIdx.x;
  int wave = tid >> 5, lane = tid & 31;
  int ml = lane & 15, sel = lane >> 4;

  v8f acc[16];
  v8f z = {0.f, 0.f, 0.f, 0.f, 0.f, 0.f, 0.f, 0.f};
#pragma unroll
  for (int i = 0; i < 16; ++i) acc[i] = z;

  // A: 16-bit 16x32 layout: lane<16 -> M=lane, K {0..7,16..23}; lane>=16 -> K {8..15,24..31}
  const _Float16* Arow = W2 + (size_t)b * OO * CC + (size_t)(o0 + wave * 16 + ml) * CC;
  const float*    xrow = x1 + (size_t)(b * NN + n0) * CC;
  int koff = sel * 16;  // B: lane<16 -> K {0..7,16..23}, lane>=16 -> K {8..15,24..31}, col N=ml

  FragH ah, ahn;
  ah.q[0] = *(const u32x4*)(Arow + sel * 8);
  ah.q[1] = *(const u32x4*)(Arow + 16 + sel * 8);

  for (int k0 = 0; k0 < CC; k0 += 32) {
    // stage V tile -> f16 (float4 loads, packed 8B LDS stores)
    for (int idx = tid; idx < 2048; idx += 512) {
      int nl = idx >> 3, k4 = (idx & 7) * 4;
      float4 v = *(const float4*)&xrow[(size_t)nl * CC + k0 + k4];
      h4v hv = {(_Float16)v.x, (_Float16)v.y, (_Float16)v.z, (_Float16)v.w};
      *(h4v*)&sb[nl][k4] = hv;
    }
    __syncthreads();

    int kn = k0 + 32;                      // prefetch next A fragment (global)
    if (kn < CC) {
      ahn.q[0] = *(const u32x4*)(Arow + kn + sel * 8);
      ahn.q[1] = *(const u32x4*)(Arow + kn + 16 + sel * 8);
    }

    FragH bhf[3];                          // B frags pipelined 2 tiles ahead
#define LOADB(dst, nt)                                                   \
    { (dst).q[0] = *(const u32x4*)(&sb[(nt) * 16 + ml][sel * 8]);        \
      (dst).q[1] = *(const u32x4*)(&sb[(nt) * 16 + ml][16 + sel * 8]); }
    LOADB(bhf[0], 0)
    LOADB(bhf[1], 1)
#pragma unroll
    for (int nt = 0; nt < 16; ++nt) {
      if (nt + 2 < 16) LOADB(bhf[(nt + 2) % 3], nt + 2)
      acc[nt] = __builtin_amdgcn_wmma_f32_16x16x32_f16(
          false, ah.v, false, bhf[nt % 3].v, (short)0, acc[nt], false, false);
    }
#undef LOADB
    __syncthreads();
    ah = ahn;
  }

  // epilogue: add bias, transpose via LDS, coalesced store to y[b,n,o]
  for (int nt = 0; nt < 16; ++nt) {
#pragma unroll
    for (int r = 0; r < 8; ++r) {
      int ol = r + sel * 8;                       // C/D layout: M = r (+8), N = ml
      tsc[wave][ml][ol] = acc[nt][r] + bias[o0 + wave * 16 + ol];
    }
    __syncthreads();
#pragma unroll
    for (int rr = 0; rr < 8; ++rr) {
      int nl2 = rr + sel * 8;
      int n_g = n0 + nt * 16 + nl2;
      y[((size_t)(b * NN + n_g)) * OO + o0 + wave * 16 + ml] = tsc[wave][nl2][ml];
    }
    __syncthreads();
  }
}

// ---------------- LayerNorm over o (in-place on d_out) -----------------------
__global__ __launch_bounds__(256) void ln_kernel(
    float* __restrict__ y, const float* __restrict__ gamma,
    const float* __restrict__ beta) {
  __shared__ float red[8], red2[8];
  size_t row = (size_t)blockIdx.x * OO;
  int tid = threadIdx.x;
  float v[8]; float s = 0.f, sq = 0.f;
#pragma unroll
  for (int i = 0; i < 8; ++i) {
    v[i] = y[row + tid + i * 256];
    s += v[i]; sq += v[i] * v[i];
  }
  for (int off = 16; off > 0; off >>= 1) {
    s  += __shfl_xor(s, off);
    sq += __shfl_xor(sq, off);
  }
  int wave = tid >> 5;
  if ((tid & 31) == 0) { red[wave] = s; red2[wave] = sq; }
  __syncthreads();
  if (tid == 0) {
    float ts = 0.f, tq = 0.f;
    for (int w = 0; w < 8; ++w) { ts += red[w]; tq += red2[w]; }
    red[0] = ts; red2[0] = tq;
  }
  __syncthreads();
  float mu   = red[0]  * (1.f / OO);
  float var  = red2[0] * (1.f / OO) - mu * mu;
  float rstd = rsqrtf(var + 1e-5f);
#pragma unroll
  for (int i = 0; i < 8; ++i) {
    int o = tid + i * 256;
    y[row + o] = (v[i] - mu) * rstd * gamma[o] + beta[o];
  }
}

// ---------------------------------------------------------------------------
extern "C" void kernel_launch(void* const* d_in, const int* in_sizes, int n_in,
                              void* d_out, int out_size, void* d_ws, size_t ws_size,
                              hipStream_t stream) {
  const float* x1     = (const float*)d_in[0];
  const float* x2     = (const float*)d_in[1];
  const float* w_proj = (const float*)d_in[2];
  const float* b_proj = (const float*)d_in[3];
  const float* gamma  = (const float*)d_in[4];
  const float* beta   = (const float*)d_in[5];
  float* out = (float*)d_out;

  // workspace layout (~37 MiB)
  char* ws = (char*)d_ws;
  float* kmax  = (float*)ws;  ws += (size_t)BB * CC * 4;                     // 16 KiB
  float* ksum  = (float*)ws;  ws += (size_t)BB * CC * 4;                     // 16 KiB
  float* qmax  = (float*)ws;  ws += (size_t)BB * HH * NN * 4;                // 512 KiB
  float* qsum  = (float*)ws;  ws += (size_t)BB * HH * NN * 4;                // 512 KiB
  float* Spart = (float*)ws;  ws += (size_t)SCHUNKS * BB * HH * CH * CH * 4; // 16 MiB
  float* S     = (float*)ws;  ws += (size_t)BB * HH * CH * CH * 4;           // 2 MiB
  _Float16* W2 = (_Float16*)ws;                                             // 16 MiB

  colstats_kernel<<<(BB * CC) / 256, 256, 0, stream>>>(x2, kmax, ksum);
  qstats_kernel<<<(BB * HH * NN) / 256, 256, 0, stream>>>(x2, qmax, qsum);
  smat_kernel<<<BB * HH * SCHUNKS, 256, 0, stream>>>(x2, kmax, ksum, qmax, qsum, Spart);
  sreduce_kernel<<<(BB * HH * CH * CH) / 256, 256, 0, stream>>>(Spart, S);
  w2_kernel<<<(BB * OO * CC) / 256, 256, 0, stream>>>(w_proj, S, W2);
  gemm_kernel<<<dim3(NN / 256, OO / 256, BB), 512, 0, stream>>>(W2, x1, b_proj, out);
  ln_kernel<<<BB * NN, 256, 0, stream>>>(out, gamma, beta);
}
